// RNNModel_78812649882436
// MI455X (gfx1250) — compile-verified, matrix operand order checked by reading, not run on hardware
//
#include <hip/hip_runtime.h>
#include <cstddef>
#include <cstdint>

// ---------------------------------------------------------------------------
// Model dims
// ---------------------------------------------------------------------------
#define SEQ   64
#define BATCH 16
#define VOCAB 32000
#define NINP  512
#define NHID  1536
#define NHL   1024
#define NEXP  8
#define ROWS  (SEQ * BATCH)          // 1024
#define LROWS (ROWS * NEXP)          // 8192

typedef __attribute__((ext_vector_type(16))) __bf16 v16bf;
typedef __attribute__((ext_vector_type(8)))  __bf16 v8bf;
typedef __attribute__((ext_vector_type(8)))  float  v8f;

// ---------------------------------------------------------------------------
// WMMA helpers (CDNA5: V_WMMA_F32_16X16X32_BF16, wave32)
// ---------------------------------------------------------------------------
__device__ __forceinline__ v8f wmma_bf16(v16bf a, v16bf b, v8f c) {
  return __builtin_amdgcn_wmma_f32_16x16x32_bf16(false, a, false, b,
                                                 (short)0, c, false, false);
}

// Load a 16x32 bf16 fragment from a row-major matrix (leading dim ld).
// Works both for A (rows = M) and for B taken from a row-major (N,K) weight
// (rows = N), per the ISA 16-bit operand lane layout:
//   lanes 0-15 : row = lane,    K = {k0+0..7, k0+16..23}
//   lanes16-31 : row = lane-16, K = {k0+8..15, k0+24..31}
__device__ __forceinline__ v16bf load_frag(const __bf16* __restrict__ p,
                                           int ld, int row0, int k0) {
  const int lane = threadIdx.x & 31;
  const int r    = lane & 15;
  const int hi   = lane >> 4;
  const __bf16* q = p + (size_t)(row0 + r) * ld + k0 + hi * 8;
  v8bf lo = *reinterpret_cast<const v8bf*>(q);
  v8bf up = *reinterpret_cast<const v8bf*>(q + 16);
  v16bf out;
#pragma unroll
  for (int i = 0; i < 8; ++i) { out[i] = lo[i]; out[i + 8] = up[i]; }
  return out;
}

// Same fragment gather, but from an LDS tile stored row-major with a padded
// row stride of 40 bf16 (80 B). 80 B stride makes the 16 ds_load_b128 starts
// hit banks {0,4,...,60}: conflict-free tiling of all 64 banks.
#define LDS_STRIDE 40
__device__ __forceinline__ v16bf load_frag_lds(const __bf16* base, int row0) {
  const int lane = threadIdx.x & 31;
  const int r    = lane & 15;
  const int hi   = lane >> 4;
  const __bf16* q = base + (row0 + r) * LDS_STRIDE + hi * 8;
  v8bf lo = *reinterpret_cast<const v8bf*>(q);
  v8bf up = *reinterpret_cast<const v8bf*>(q + 16);
  v16bf out;
#pragma unroll
  for (int i = 0; i < 8; ++i) { out[i] = lo[i]; out[i + 8] = up[i]; }
  return out;
}

// ---------------------------------------------------------------------------
// Utility kernels
// ---------------------------------------------------------------------------
__global__ void k_cast_bf16(const float* __restrict__ in,
                            __bf16* __restrict__ out, int n) {
  int i = blockIdx.x * blockDim.x + threadIdx.x;
  int stride = gridDim.x * blockDim.x;
  for (; i < n; i += stride) out[i] = (__bf16)in[i];
}

__global__ void k_copy_f32(const float* __restrict__ in,
                           float* __restrict__ out, int n) {
  int i = blockIdx.x * blockDim.x + threadIdx.x;
  int stride = gridDim.x * blockDim.x;
  for (; i < n; i += stride) out[i] = in[i];
}

__global__ void k_zero_f32(float* __restrict__ p, int n) {
  int i = blockIdx.x * blockDim.x + threadIdx.x;
  if (i < n) p[i] = 0.f;
}

// Embedding gather + cast to bf16: x0[row, :] = emb[tokens[row], :]
__global__ void k_embed(const int* __restrict__ tokens,
                        const float* __restrict__ emb,
                        __bf16* __restrict__ x0) {
  const int row = blockIdx.x;
  const int tok = tokens[row];
  const float* src = emb + (size_t)tok * NINP;
  __bf16* dst = x0 + (size_t)row * NINP;
  for (int i = threadIdx.x; i < NINP; i += blockDim.x) dst[i] = (__bf16)src[i];
}

// ---------------------------------------------------------------------------
// Generic bf16 WMMA GEMM: C[M,N] = A[M,K] @ W[N,K]^T + bias[N]
// block = 128 threads (4 waves), each wave owns a 32x32 sub-tile of a 64x64
// block tile. MODE 0: store f32.  MODE 1: store bf16 of tanh(x)*gain[n%NINP]
// (latent epilogue).
// ---------------------------------------------------------------------------
template <int MODE>
__global__ void k_gemm(const __bf16* __restrict__ A,
                       const __bf16* __restrict__ W,
                       const float* __restrict__ bias,
                       const float* __restrict__ gain,
                       void* __restrict__ Cout, int N, int K) {
  const int wave = threadIdx.x >> 5;
  const int n0 = blockIdx.x * 64 + (wave & 1) * 32;
  const int m0 = blockIdx.y * 64 + (wave >> 1) * 32;

  v8f acc[2][2] = {};
  for (int k0 = 0; k0 < K; k0 += 32) {
    v16bf a0 = load_frag(A, K, m0,      k0);
    v16bf a1 = load_frag(A, K, m0 + 16, k0);
    v16bf b0 = load_frag(W, K, n0,      k0);
    v16bf b1 = load_frag(W, K, n0 + 16, k0);
    acc[0][0] = wmma_bf16(a0, b0, acc[0][0]);
    acc[0][1] = wmma_bf16(a0, b1, acc[0][1]);
    acc[1][0] = wmma_bf16(a1, b0, acc[1][0]);
    acc[1][1] = wmma_bf16(a1, b1, acc[1][1]);
  }

  const int lane = threadIdx.x & 31;
  const int r = lane & 15;
  const int hi = lane >> 4;
#pragma unroll
  for (int mi = 0; mi < 2; ++mi)
#pragma unroll
    for (int ni = 0; ni < 2; ++ni) {
#pragma unroll
      for (int v = 0; v < 8; ++v) {
        const int m = m0 + mi * 16 + v + 8 * hi;
        const int n = n0 + ni * 16 + r;
        float x = acc[mi][ni][v] + bias[n];
        if constexpr (MODE == 0) {
          reinterpret_cast<float*>(Cout)[(size_t)m * N + n] = x;
        } else {
          reinterpret_cast<__bf16*>(Cout)[(size_t)m * N + n] =
              (__bf16)(tanhf(x) * gain[n & (NINP - 1)]);
        }
      }
    }
}

// ---------------------------------------------------------------------------
// One LSTM time step. One wave (32 threads) per 16 hidden units: computes the
// i/f/g/o gate tiles g = xg[t] + h @ Whh^T with 4 WMMA accumulators sharing a
// single A fragment, then the cell nonlinearity fully in registers.
// ---------------------------------------------------------------------------
__global__ void k_lstm_step(const float* __restrict__ xg,     // (ROWS, 4H)
                            const __bf16* __restrict__ h_in,  // (B, H) bf16
                            const __bf16* __restrict__ Whh,   // (4H, H) bf16
                            float* __restrict__ c,            // (B, H) f32
                            __bf16* __restrict__ h_out,       // (B, H) bf16
                            __bf16* __restrict__ xseq,        // (ROWS, H)
                            int H, int t) {
  const int j0 = blockIdx.x * 16;
  v8f acc[4] = {};
  for (int k0 = 0; k0 < H; k0 += 32) {
    v16bf a = load_frag(h_in, H, 0, k0);  // rows = batch (M=16)
#pragma unroll
    for (int g = 0; g < 4; ++g) {
      v16bf b = load_frag(Whh, H, g * H + j0, k0);
      acc[g] = wmma_bf16(a, b, acc[g]);
    }
  }

  const int lane = threadIdx.x & 31;
  const int r = lane & 15;
  const int hi = lane >> 4;
  const int j = j0 + r;
#pragma unroll
  for (int v = 0; v < 8; ++v) {
    const int b_ = v + 8 * hi;
    const float* xr = xg + (size_t)(t * BATCH + b_) * 4 * H;
    float gi = acc[0][v] + xr[j];
    float gf = acc[1][v] + xr[H + j];
    float gg = acc[2][v] + xr[2 * H + j];
    float go = acc[3][v] + xr[3 * H + j];
    float si = 1.f / (1.f + __expf(-gi));
    float sf = 1.f / (1.f + __expf(-gf));
    float so = 1.f / (1.f + __expf(-go));
    float cn = sf * c[b_ * H + j] + si * tanhf(gg);
    float hn = so * tanhf(cn);
    c[b_ * H + j] = cn;
    h_out[b_ * H + j] = (__bf16)hn;
    xseq[(size_t)(t * BATCH + b_) * H + j] = (__bf16)hn;
  }
}

// ---------------------------------------------------------------------------
// Prior logits: priorlog[n*8+e] = dot(out[n,:NHL], prior_W[e,:]).
// One wave per (n,e).
// ---------------------------------------------------------------------------
__global__ void k_prior(const __bf16* __restrict__ x3,
                        const float* __restrict__ prior_W,
                        float* __restrict__ priorlog) {
  const int idx = blockIdx.x;           // n*8+e
  const int n = idx >> 3, e = idx & 7;
  const int lane = threadIdx.x;
  float s = 0.f;
  for (int h = lane; h < NHL; h += 32)
    s += (float)x3[(size_t)n * NHL + h] * prior_W[(size_t)e * NHL + h];
#pragma unroll
  for (int m = 16; m; m >>= 1) s += __shfl_xor(s, m, 32);
  if (lane == 0) priorlog[idx] = s;
}

// coeff[n,e] = softmax_e(priorlog[n,:]) / Ssum[n,e]
__global__ void k_coeff(const float* __restrict__ priorlog,
                        const float* __restrict__ Ssum,
                        float* __restrict__ coeff) {
  const int n = blockIdx.x * blockDim.x + threadIdx.x;
  if (n >= ROWS) return;
  float v[NEXP], mx = -1e30f;
#pragma unroll
  for (int e = 0; e < NEXP; ++e) { v[e] = priorlog[n * NEXP + e]; mx = fmaxf(mx, v[e]); }
  float s = 0.f;
#pragma unroll
  for (int e = 0; e < NEXP; ++e) { v[e] = __expf(v[e] - mx); s += v[e]; }
#pragma unroll
  for (int e = 0; e < NEXP; ++e)
    coeff[n * NEXP + e] = v[e] / (s * Ssum[n * NEXP + e]);
}

// ---------------------------------------------------------------------------
// Fused decoder. logit[m, v] = latent[m,:512] @ decW[v,:512]^T + dec_b[v].
// The 128-column decW tile is shared by all 4 waves of the block, so it is
// staged once into LDS with async copies (global_load_async_to_lds_b128,
// ASYNCcnt-tracked, double-buffered) and consumed with conflict-free
// ds_load_b128 fragment gathers.
// PASS1: Ssum[m] += sum_v exp(logit[m,v])          (atomic row sums)
// PASS2: out[n, v] = log(sum_e coeff[n*8+e] * exp(logit[n*8+e, v]) + 1e-8)
// 16-row M-tiles => the 8 accumulator regs of a lane are exactly the 8
// experts of one output row: the mixture reduction is register-local.
// ---------------------------------------------------------------------------
#define KSTEPS (NINP / 32)   // 16

template <bool PASS2>
__global__ void k_dec(const __bf16* __restrict__ latent,   // (LROWS, 512)
                      const __bf16* __restrict__ decW,     // (VOCAB, 512)
                      const float* __restrict__ dec_b,
                      const float* __restrict__ coeff,
                      float* __restrict__ SorOut) {
  __shared__ __bf16 smem[2][128 * LDS_STRIDE];

  const int tid  = threadIdx.x;
  const int wave = tid >> 5;
  const int m0 = (blockIdx.y * 4 + wave) * 16;
  const int c0 = blockIdx.x * 128;

  // Stage decW[c0..c0+127, k0..k0+31] (8 KB) into smem[buf]: 512 16-byte
  // chunks, 4 per thread, coalesced 64 B per weight row.
  auto stage = [&](int buf, int k0) {
#pragma unroll
    for (int i = 0; i < 4; ++i) {
      const int chunk = tid + i * 128;        // 0..511
      const int row = chunk >> 2;             // 0..127
      const int cir = chunk & 3;              // 16B chunk within 64B row
      const uint64_t g = (uint64_t)(uintptr_t)(decW + (size_t)(c0 + row) * NINP + k0)
                         + (uint64_t)cir * 16;
      const uint32_t l = (uint32_t)(uintptr_t)(&smem[buf][row * LDS_STRIDE])
                         + (uint32_t)cir * 16;
      asm volatile("global_load_async_to_lds_b128 %0, %1, off"
                   :: "v"(l), "v"(g) : "memory");
    }
  };

  v8f acc[8] = {};
  stage(0, 0);
  for (int ks = 0; ks < KSTEPS; ++ks) {
    const int k0 = ks * 32;
    if (ks + 1 < KSTEPS) {
      stage((ks + 1) & 1, k0 + 32);
      asm volatile("s_wait_asynccnt 0x4" ::: "memory");  // prev stage done
    } else {
      asm volatile("s_wait_asynccnt 0x0" ::: "memory");
    }
    __syncthreads();                                     // tile visible to all
    v16bf a = load_frag(latent, NINP, m0, k0);
    const __bf16* bbase = smem[ks & 1];
#pragma unroll
    for (int nt = 0; nt < 8; ++nt) {
      v16bf b = load_frag_lds(bbase, nt * 16);
      acc[nt] = wmma_bf16(a, b, acc[nt]);
    }
    __syncthreads();             // done reading before this buffer is reused
  }

  const int lane = tid & 31;
  const int r = lane & 15;
  const int hi = lane >> 4;
  if (!PASS2) {
#pragma unroll
    for (int v = 0; v < 8; ++v) {
      float s = 0.f;
#pragma unroll
      for (int nt = 0; nt < 8; ++nt)
        s += __expf(acc[nt][v] + dec_b[c0 + nt * 16 + r]);
      // reduce over the 16 columns held by each half-wave group
      s += __shfl_xor(s, 1, 32);
      s += __shfl_xor(s, 2, 32);
      s += __shfl_xor(s, 4, 32);
      s += __shfl_xor(s, 8, 32);
      if (r == 0) atomicAdd(&SorOut[m0 + v + 8 * hi], s);
    }
  } else {
    float cf[8];
#pragma unroll
    for (int v = 0; v < 8; ++v) cf[v] = coeff[m0 + 8 * hi + v];
    const int n = (m0 >> 3) + hi;   // output row (0..1023)
#pragma unroll
    for (int nt = 0; nt < 8; ++nt) {
      float mix = 0.f;
#pragma unroll
      for (int v = 0; v < 8; ++v)
        mix += cf[v] * __expf(acc[nt][v] + dec_b[c0 + nt * 16 + r]);
      SorOut[(size_t)n * VOCAB + c0 + nt * 16 + r] = __logf(mix + 1e-8f);
    }
  }
}

// ---------------------------------------------------------------------------
// Host-side driver
// ---------------------------------------------------------------------------
static inline int cdiv(int a, int b) { return (a + b - 1) / b; }

extern "C" void kernel_launch(void* const* d_in, const int* in_sizes, int n_in,
                              void* d_out, int out_size, void* d_ws, size_t ws_size,
                              hipStream_t stream) {
  (void)in_sizes; (void)n_in; (void)out_size; (void)ws_size;

  const int*   tokens  = (const int*)  d_in[0];
  const float* h0      = (const float*)d_in[1];
  const float* c0      = (const float*)d_in[2];
  const float* h1      = (const float*)d_in[3];
  const float* c1      = (const float*)d_in[4];
  const float* h2      = (const float*)d_in[5];
  const float* c2      = (const float*)d_in[6];
  const float* emb     = (const float*)d_in[7];
  const float* Wih0    = (const float*)d_in[8];
  const float* Whh0    = (const float*)d_in[9];
  const float* b0      = (const float*)d_in[10];
  const float* Wih1    = (const float*)d_in[11];
  const float* Whh1    = (const float*)d_in[12];
  const float* b1      = (const float*)d_in[13];
  const float* Wih2    = (const float*)d_in[14];
  const float* Whh2    = (const float*)d_in[15];
  const float* b2      = (const float*)d_in[16];
  const float* prior_W = (const float*)d_in[17];
  const float* latW    = (const float*)d_in[18];
  const float* lat_b   = (const float*)d_in[19];
  const float* dec_W   = (const float*)d_in[20];
  const float* dec_b   = (const float*)d_in[21];
  const float* gain    = (const float*)d_in[22];
  float* out = (float*)d_out;

  // ---- workspace carve-up ----
  char* wsb = (char*)d_ws;
  size_t off = 0;
  auto alloc = [&](size_t bytes) -> void* {
    off = (off + 255) & ~(size_t)255;
    void* p = wsb + off;
    off += bytes;
    return p;
  };
  __bf16* decW_bf = (__bf16*)alloc((size_t)VOCAB * NINP * 2);
  __bf16* Wih0_bf = (__bf16*)alloc((size_t)4 * NHID * NINP * 2);
  __bf16* Whh0_bf = (__bf16*)alloc((size_t)4 * NHID * NHID * 2);
  __bf16* Wih1_bf = (__bf16*)alloc((size_t)4 * NHID * NHID * 2);
  __bf16* Whh1_bf = (__bf16*)alloc((size_t)4 * NHID * NHID * 2);
  __bf16* Wih2_bf = (__bf16*)alloc((size_t)4 * NHL * NHID * 2);
  __bf16* Whh2_bf = (__bf16*)alloc((size_t)4 * NHL * NHL * 2);
  __bf16* latW_bf = (__bf16*)alloc((size_t)NEXP * NINP * NHL * 2);
  __bf16* x0_bf   = (__bf16*)alloc((size_t)ROWS * NINP * 2);
  __bf16* x1_bf   = (__bf16*)alloc((size_t)ROWS * NHID * 2);
  __bf16* x2_bf   = (__bf16*)alloc((size_t)ROWS * NHID * 2);
  __bf16* x3_bf   = (__bf16*)alloc((size_t)ROWS * NHL * 2);
  float*  xg      = (float*) alloc((size_t)ROWS * 4 * NHID * 4);
  __bf16* lat_bf  = (__bf16*)alloc((size_t)LROWS * NINP * 2);
  __bf16* h_ping  = (__bf16*)alloc((size_t)BATCH * NHID * 2);
  __bf16* h_pong  = (__bf16*)alloc((size_t)BATCH * NHID * 2);
  float*  cbuf    = (float*) alloc((size_t)BATCH * NHID * 4);
  float*  Ssum    = (float*) alloc((size_t)LROWS * 4);
  float*  priorlg = (float*) alloc((size_t)LROWS * 4);
  float*  coeff   = (float*) alloc((size_t)LROWS * 4);

  auto cast = [&](const float* src, __bf16* dst, size_t n) {
    int blocks = (int)((n + 255) / 256);
    if (blocks > 4096) blocks = 4096;
    k_cast_bf16<<<blocks, 256, 0, stream>>>(src, dst, (int)n);
  };

  // ---- weight conversions (each call; deterministic) ----
  cast(dec_W, decW_bf, (size_t)VOCAB * NINP);
  cast(Wih0,  Wih0_bf, (size_t)4 * NHID * NINP);
  cast(Whh0,  Whh0_bf, (size_t)4 * NHID * NHID);
  cast(Wih1,  Wih1_bf, (size_t)4 * NHID * NHID);
  cast(Whh1,  Whh1_bf, (size_t)4 * NHID * NHID);
  cast(Wih2,  Wih2_bf, (size_t)4 * NHL * NHID);
  cast(Whh2,  Whh2_bf, (size_t)4 * NHL * NHL);
  cast(latW,  latW_bf, (size_t)NEXP * NINP * NHL);

  // ---- embedding ----
  k_embed<<<ROWS, 128, 0, stream>>>(tokens, emb, x0_bf);

  // ---- LSTM layers ----
  auto lstm_layer = [&](const __bf16* xin, int Din, const __bf16* WihB,
                        const __bf16* WhhB, const float* bias,
                        const float* h_init, const float* c_init, int H,
                        __bf16* xout) {
    // batched input projection: xg = xin @ Wih^T + b  (M=ROWS, N=4H, K=Din)
    k_gemm<0><<<dim3(4 * H / 64, ROWS / 64), 128, 0, stream>>>(
        xin, WihB, bias, nullptr, xg, 4 * H, Din);
    cast(h_init, h_ping, (size_t)BATCH * H);
    {
      int n = BATCH * H;
      k_copy_f32<<<cdiv(n, 256), 256, 0, stream>>>(c_init, cbuf, n);
    }
    __bf16* hin = h_ping;
    __bf16* hout = h_pong;
    for (int t = 0; t < SEQ; ++t) {
      k_lstm_step<<<H / 16, 32, 0, stream>>>(xg, hin, WhhB, cbuf, hout, xout,
                                             H, t);
      __bf16* tmp = hin; hin = hout; hout = tmp;
    }
  };

  lstm_layer(x0_bf, NINP, Wih0_bf, Whh0_bf, b0, h0, c0, NHID, x1_bf);
  lstm_layer(x1_bf, NHID, Wih1_bf, Whh1_bf, b1, h1, c1, NHID, x2_bf);
  lstm_layer(x2_bf, NHID, Wih2_bf, Whh2_bf, b2, h2, c2, NHL,  x3_bf);

  // ---- MoS head ----
  // latent = bf16( tanh(out @ latW^T + lat_b) * gain )  (M=ROWS, N=E*NINP)
  k_gemm<1><<<dim3(NEXP * NINP / 64, ROWS / 64), 128, 0, stream>>>(
      x3_bf, latW_bf, lat_b, gain, lat_bf, NEXP * NINP, NHL);

  // prior logits + row-sum buffer
  k_prior<<<LROWS, 32, 0, stream>>>(x3_bf, prior_W, priorlg);
  k_zero_f32<<<cdiv(LROWS, 256), 256, 0, stream>>>(Ssum, LROWS);

  // decoder pass 1: per-expert-row exp sums
  k_dec<false><<<dim3(VOCAB / 128, LROWS / 64), 128, 0, stream>>>(
      lat_bf, decW_bf, dec_b, nullptr, Ssum);

  // coeff = softmax(prior) / Ssum
  k_coeff<<<cdiv(ROWS, 256), 256, 0, stream>>>(priorlg, Ssum, coeff);

  // decoder pass 2: recompute logits, mix experts in-register, emit log-probs
  k_dec<true><<<dim3(VOCAB / 128, LROWS / 64), 128, 0, stream>>>(
      lat_bf, decW_bf, dec_b, coeff, out);
}
